// LinearCrossAttention_51256139711131
// MI455X (gfx1250) — matrix operand order repaired and checked by reading, not compiled, not run on hardware
//
#include <hip/hip_runtime.h>
#include <hip/hip_bf16.h>
#include <math.h>

#define BDIM 8
#define NDIM 4096
#define CDIM 512
#define TDIM 4096
#define HDIM 8
#define DHEAD 64
#define NTERM 3

typedef __attribute__((ext_vector_type(16))) __bf16 v16bf;
typedef __attribute__((ext_vector_type(8)))  __bf16 v8bf;
typedef __attribute__((ext_vector_type(4)))  __bf16 v4bf;
typedef __attribute__((ext_vector_type(8)))  float  v8f;

// ---------------------------------------------------------------------------
// gfx1250 async global->LDS copy (ASYNCcnt-tracked DMA path, ISA ch.10/15.18).
// lds_off is the wave-relative LDS byte address == low 32 bits of the generic
// pointer (ISA 10.2).
// ---------------------------------------------------------------------------
__device__ __forceinline__ void async_copy_b128(unsigned lds_off, const void* g) {
  asm volatile("global_load_async_to_lds_b128 %0, %1, off"
               :: "v"(lds_off), "v"(g)
               : "memory");
}
__device__ __forceinline__ void wait_asynccnt0() {
  asm volatile("s_wait_asynccnt 0x0" ::: "memory");
}
__device__ __forceinline__ unsigned lds_off_of(const void* p) {
  return (unsigned)(unsigned long long)p;
}

// ---------------------------------------------------------------------------
// Fragment loaders (wave32, v_wmma_f32_16x16x32_bf16 operand layouts,
// CDNA5 ISA 7.12.2):
//  A 16x32 bf16: lane L holds row m=L&15; element e -> k = 16*(e>>3) + 8*(L>>4) + (e&7)
//  B 32x16 bf16 (held transposed [n][k] row-major in LDS):
//    lane L holds col n=L&15; element e -> k = 16*(e>>3) + 8*(L>>4) + (e&7)
// Both are two contiguous 16B LDS reads per lane (ds_load_b128 pairs).
// ---------------------------------------------------------------------------
__device__ __forceinline__ v16bf frag_a_rowmajor(const __bf16* base, int rowStride, int row0) {
  const int lane = threadIdx.x & 31;
  const __bf16* p = base + (size_t)(row0 + (lane & 15)) * rowStride + ((lane >> 4) << 3);
  v8bf lo = *(const v8bf*)(p);
  v8bf hi = *(const v8bf*)(p + 16);
  v16bf r;
#pragma unroll
  for (int j = 0; j < 8; ++j) { r[j] = lo[j]; r[8 + j] = hi[j]; }
  return r;
}

__device__ __forceinline__ v16bf frag_b_colmajor(const __bf16* base, int rowStride, int col0) {
  const int lane = threadIdx.x & 31;
  const __bf16* p = base + (size_t)(col0 + (lane & 15)) * rowStride + ((lane >> 4) << 3);
  v8bf lo = *(const v8bf*)(p);
  v8bf hi = *(const v8bf*)(p + 16);
  v16bf r;
#pragma unroll
  for (int j = 0; j < 8; ++j) { r[j] = lo[j]; r[8 + j] = hi[j]; }
  return r;
}

#define WMMA_BF16(A, B, C) \
  __builtin_amdgcn_wmma_f32_16x16x32_bf16(false, (A), false, (B), (short)0, (C), false, false)

// ---------------------------------------------------------------------------
// Projection GEMM: out[m][n] = sum_k A[m][k] * W[n][k] + bias[n]
// Block tile 128x64, 8 waves (4 row groups x 2 col groups), each wave a
// 32x32 sub-tile (2x2 accumulators, 4 WMMA per K-step from 8 ds_load_b128).
// Double-buffered LDS. fp32 inputs: register-staged + converted to bf16.
// bf16 inputs (final projection): A tile copied by the async global->LDS DMA,
// waited with s_wait_asynccnt only after the WMMAs of the current step.
// MODE: 0 = bf16 out, 1 = softmax over the 64-col tile (== one head) then
// bf16 out, 2 = fp32 out.
// ---------------------------------------------------------------------------
template <int MODE, bool ABF16>
__global__ __launch_bounds__(256)
void proj_kernel(const void* __restrict__ Ap, const float* __restrict__ W,
                 const float* __restrict__ bias, void* __restrict__ outp,
                 int M, int K, int Nc) {
  __shared__ __align__(16) __bf16 As[2][128][32];
  __shared__ __align__(16) __bf16 Bs[2][64][32];

  const int m0 = blockIdx.x * 128;
  const int n0 = blockIdx.y * 64;
  const int tid = threadIdx.x;
  const int wid = tid >> 5;
  const int wr = (wid >> 1) << 5;  // 0,32,64,96
  const int wc = (wid & 1) << 5;   // 0,32

  const float* Af = (const float*)Ap;
  const __bf16* Ab = (const __bf16*)Ap;
  const unsigned ldsAs = lds_off_of(&As[0][0][0]);

  float ar[4][4];  // staged A (fp32 path): 4 vec4 per thread (128x32 tile)
  float br[2][4];  // staged W: 2 vec4 per thread (64x32 tile)

  auto loadTile = [&](int k0, int pdst) {
    if (ABF16) {
      // A tile is bf16 row-major in global: straight 8KB copy -> async DMA.
#pragma unroll
      for (int j = 0; j < 2; ++j) {
        int q = (j << 8) + tid;              // 0..511 16-byte chunks
        int r = q >> 2, c8 = (q & 3) << 3;   // 4 chunks per 32-elem row
        async_copy_b128(ldsAs + (unsigned)pdst * 8192u + (unsigned)((r * 32 + c8) << 1),
                        Ab + (size_t)(m0 + r) * K + k0 + c8);
      }
    } else {
#pragma unroll
      for (int j = 0; j < 4; ++j) {
        int v = (j << 8) + tid;              // 0..1023 vec4 slots
        int r = v >> 3, c = (v & 7) << 2;
        float4 t = *(const float4*)(Af + (size_t)(m0 + r) * K + k0 + c);
        ar[j][0] = t.x; ar[j][1] = t.y; ar[j][2] = t.z; ar[j][3] = t.w;
      }
    }
#pragma unroll
    for (int j = 0; j < 2; ++j) {
      int v = (j << 8) + tid;
      int r = v >> 3, c = (v & 7) << 2;
      float4 t = *(const float4*)(W + (size_t)(n0 + r) * K + k0 + c);
      br[j][0] = t.x; br[j][1] = t.y; br[j][2] = t.z; br[j][3] = t.w;
    }
  };
  auto storeTile = [&](int p) {
    if (!ABF16) {
#pragma unroll
      for (int j = 0; j < 4; ++j) {
        int v = (j << 8) + tid;
        int r = v >> 3, c = (v & 7) << 2;
        v4bf t = { (__bf16)ar[j][0], (__bf16)ar[j][1], (__bf16)ar[j][2], (__bf16)ar[j][3] };
        *(v4bf*)&As[p][r][c] = t;
      }
    }
#pragma unroll
    for (int j = 0; j < 2; ++j) {
      int v = (j << 8) + tid;
      int r = v >> 3, c = (v & 7) << 2;
      v4bf t = { (__bf16)br[j][0], (__bf16)br[j][1], (__bf16)br[j][2], (__bf16)br[j][3] };
      *(v4bf*)&Bs[p][r][c] = t;
    }
  };

  v8f acc00 = {}, acc01 = {}, acc10 = {}, acc11 = {};

  loadTile(0, 0);
  storeTile(0);
  if (ABF16) wait_asynccnt0();
  __syncthreads();
  const int nsteps = K >> 5;
  for (int s = 0; s < nsteps; ++s) {
    int p = s & 1;
    bool hasNext = (s + 1 < nsteps);
    if (hasNext) loadTile((s + 1) << 5, p ^ 1);  // overlaps with WMMAs below
    v16bf a0 = frag_a_rowmajor(&As[p][0][0], 32, wr);
    v16bf a1 = frag_a_rowmajor(&As[p][0][0], 32, wr + 16);
    v16bf b0 = frag_b_colmajor(&Bs[p][0][0], 32, wc);
    v16bf b1 = frag_b_colmajor(&Bs[p][0][0], 32, wc + 16);
    acc00 = WMMA_BF16(a0, b0, acc00);
    acc01 = WMMA_BF16(a0, b1, acc01);
    acc10 = WMMA_BF16(a1, b0, acc10);
    acc11 = WMMA_BF16(a1, b1, acc11);
    if (hasNext) {
      storeTile(p ^ 1);
      if (ABF16) wait_asynccnt0();
    }
    __syncthreads();
  }

  // C/D layout: VGPR r -> m = r + 8*(lane>>4); n = lane&15
  const int lane = tid & 31;
  const int nl = lane & 15;
  const int mb = (lane >> 4) << 3;
  const float bv0 = bias[n0 + wc + nl];
  const float bv1 = bias[n0 + wc + nl + 16];

  if constexpr (MODE == 1) {
    __shared__ float Cs[128][64];
#pragma unroll
    for (int r = 0; r < 8; ++r) {
      Cs[wr + mb + r][wc + nl]           = acc00[r] + bv0;
      Cs[wr + mb + r][wc + nl + 16]      = acc01[r] + bv1;
      Cs[wr + 16 + mb + r][wc + nl]      = acc10[r] + bv0;
      Cs[wr + 16 + mb + r][wc + nl + 16] = acc11[r] + bv1;
    }
    __syncthreads();
    if (tid < 128) {  // softmax over the head dimension (this 64-wide tile)
      float mx = -1e30f;
      for (int c = 0; c < 64; ++c) mx = fmaxf(mx, Cs[tid][c]);
      float s = 0.f;
      for (int c = 0; c < 64; ++c) s += __expf(Cs[tid][c] - mx);
      float inv = 1.f / s;
      __bf16* out = (__bf16*)outp;
      for (int c = 0; c < 64; ++c)
        out[(size_t)(m0 + tid) * Nc + n0 + c] = (__bf16)(__expf(Cs[tid][c] - mx) * inv);
    }
  } else if constexpr (MODE == 0) {
    __bf16* out = (__bf16*)outp;
#pragma unroll
    for (int r = 0; r < 8; ++r) {
      size_t row0 = (size_t)(m0 + wr + mb + r) * Nc + n0;
      size_t row1 = (size_t)(m0 + wr + 16 + mb + r) * Nc + n0;
      out[row0 + wc + nl]      = (__bf16)(acc00[r] + bv0);
      out[row0 + wc + nl + 16] = (__bf16)(acc01[r] + bv1);
      out[row1 + wc + nl]      = (__bf16)(acc10[r] + bv0);
      out[row1 + wc + nl + 16] = (__bf16)(acc11[r] + bv1);
    }
  } else {
    float* out = (float*)outp;
#pragma unroll
    for (int r = 0; r < 8; ++r) {
      size_t row0 = (size_t)(m0 + wr + mb + r) * Nc + n0;
      size_t row1 = (size_t)(m0 + wr + 16 + mb + r) * Nc + n0;
      out[row0 + wc + nl]      = acc00[r] + bv0;
      out[row0 + wc + nl + 16] = acc01[r] + bv1;
      out[row1 + wc + nl]      = acc10[r] + bv0;
      out[row1 + wc + nl + 16] = acc11[r] + bv1;
    }
  }
}

// ---------------------------------------------------------------------------
// ksum[i*B+b][c] = sum_t k_bf16[(i*B+b)*T + t][c]
// ---------------------------------------------------------------------------
__global__ __launch_bounds__(256)
void ksum_kernel(const __bf16* __restrict__ kbuf, float* __restrict__ ksum) {
  int idx = blockIdx.x * blockDim.x + threadIdx.x;  // over NTERM*B*C
  int c = idx % CDIM;
  int ib = idx / CDIM;
  const __bf16* p = kbuf + (size_t)ib * TDIM * CDIM + c;
  float s = 0.f;
  for (int t = 0; t < TDIM; ++t) {
    if ((t & 31) == 0 && t + 32 < TDIM)
      __builtin_prefetch(p + (size_t)(t + 32) * CDIM, 0, 0);  // global_prefetch_b8
    s += (float)p[(size_t)t * CDIM];
  }
  ksum[idx] = s;
}

// ---------------------------------------------------------------------------
// ctx[i,b,h] (64x64 f32) = k^T @ v over T=4096. One workgroup per (i,b,h).
// k/v tiles staged transposed ([d][t]) so both fragments load contiguously.
// ---------------------------------------------------------------------------
__global__ __launch_bounds__(256)
void ctx_kernel(const __bf16* __restrict__ kbuf, const __bf16* __restrict__ vbuf,
                float* __restrict__ ctx) {
  __shared__ __align__(16) __bf16 KsT[64][32];
  __shared__ __align__(16) __bf16 VsT[64][32];
  const int g = blockIdx.x;  // 0..NTERM*B*H-1
  const int b = (g / HDIM) % BDIM;
  const int i = g / (BDIM * HDIM);
  const int h = g % HDIM;
  const size_t base = (size_t)(i * BDIM + b) * TDIM * CDIM + h * DHEAD;
  const int tid = threadIdx.x;
  const int wid = tid >> 5;
  const int wr = (wid >> 1) << 4;
  const int wc = (wid & 1) << 5;
  v8f acc0 = {};
  v8f acc1 = {};
  for (int t0 = 0; t0 < TDIM; t0 += 32) {
#pragma unroll
    for (int j = 0; j < 8; ++j) {
      int idx = (j << 8) + tid;
      int t = idx >> 6, dd = idx & 63;
      KsT[dd][t] = kbuf[base + (size_t)(t0 + t) * CDIM + dd];
      VsT[dd][t] = vbuf[base + (size_t)(t0 + t) * CDIM + dd];
    }
    __syncthreads();
    v16bf a  = frag_a_rowmajor(&KsT[0][0], 32, wr);        // A[m=d][k=t]
    v16bf b0 = frag_b_colmajor(&VsT[0][0], 32, wc);        // B[k=t][n=e]
    v16bf b1 = frag_b_colmajor(&VsT[0][0], 32, wc + 16);
    acc0 = WMMA_BF16(a, b0, acc0);
    acc1 = WMMA_BF16(a, b1, acc1);
    __syncthreads();
  }
  const int lane = tid & 31;
  const int nl = wc + (lane & 15);
  const int mb = wr + ((lane >> 4) << 3);
  float* cp = ctx + (size_t)g * DHEAD * DHEAD;
#pragma unroll
  for (int r = 0; r < 8; ++r) {
    cp[(size_t)(mb + r) * DHEAD + nl]      = acc0[r];
    cp[(size_t)(mb + r) * DHEAD + nl + 16] = acc1[r];
  }
}

// ---------------------------------------------------------------------------
// attn[b, n, h*64+e] = q + sum_i (q @ ctx_i) * D_inv_i, bf16 out.
// Qs tile staged by the async global->LDS DMA (byte-exact copy).
// ---------------------------------------------------------------------------
__global__ __launch_bounds__(256)
void attn_kernel(const __bf16* __restrict__ qbuf, const float* __restrict__ ctx,
                 const float* __restrict__ ksum, __bf16* __restrict__ attn) {
  __shared__ __align__(16) __bf16 Qs[64][64];
  __shared__ __align__(16) __bf16 CtxT[NTERM][64][64];
  __shared__ float ksum_s[NTERM][64];
  __shared__ float dinv_s[NTERM][64];

  const int m0 = blockIdx.x * 64;
  const int b = blockIdx.y / HDIM;
  const int h = blockIdx.y % HDIM;
  const int tid = threadIdx.x;

  const __bf16* qb = qbuf + ((size_t)b * NDIM + m0) * CDIM + h * DHEAD;
  const unsigned ldsQs = lds_off_of(&Qs[0][0]);
#pragma unroll
  for (int j = 0; j < 2; ++j) {
    int q = (j << 8) + tid;             // 0..511 16-byte chunks of the 8KB tile
    int r = q >> 3, c8 = (q & 7) << 3;  // 8 chunks per 64-elem row
    async_copy_b128(ldsQs + (unsigned)((r * 64 + c8) << 1),
                    qb + (size_t)r * CDIM + c8);
  }
#pragma unroll
  for (int i = 0; i < NTERM; ++i) {
    const float* cg = ctx + (size_t)((i * BDIM + b) * HDIM + h) * DHEAD * DHEAD;
#pragma unroll
    for (int j = 0; j < 16; ++j) {
      int idx = (j << 8) + tid;
      int kk = idx >> 6, n = idx & 63;
      CtxT[i][n][kk] = (__bf16)cg[idx];  // cg row-major [kk][n] -> transposed stage
    }
  }
  if (tid < NTERM * 64) {
    int i = tid >> 6, dd = tid & 63;
    ksum_s[i][dd] = ksum[(size_t)(i * BDIM + b) * CDIM + h * DHEAD + dd];
  }
  wait_asynccnt0();
  __syncthreads();
  if (tid < NTERM * 64) {
    int i = tid >> 6, r = tid & 63;
    float s = 0.f;
    for (int dd = 0; dd < 64; ++dd) s += (float)Qs[r][dd] * ksum_s[i][dd];
    dinv_s[i][r] = 1.0f / s;
  }
  __syncthreads();

  const int wid = tid >> 5;
  const int wr = (wid >> 1) << 4;
  const int wc = (wid & 1) << 5;
  v8f acc[NTERM][2] = {};
#pragma unroll
  for (int kc = 0; kc < 64; kc += 32) {
    v16bf a = frag_a_rowmajor(&Qs[0][kc], 64, wr);
#pragma unroll
    for (int i = 0; i < NTERM; ++i) {
      v16bf b0 = frag_b_colmajor(&CtxT[i][0][kc], 64, wc);
      v16bf b1 = frag_b_colmajor(&CtxT[i][0][kc], 64, wc + 16);
      acc[i][0] = WMMA_BF16(a, b0, acc[i][0]);
      acc[i][1] = WMMA_BF16(a, b1, acc[i][1]);
    }
  }
  const int lane = tid & 31;
  const int nl = wc + (lane & 15);
  const int mb = wr + ((lane >> 4) << 3);
  __bf16* op = attn + ((size_t)b * NDIM + m0) * CDIM + h * DHEAD;
#pragma unroll
  for (int r = 0; r < 8; ++r) {
    int m = mb + r;
    float o0 = (float)Qs[m][nl];
    float o1 = (float)Qs[m][nl + 16];
#pragma unroll
    for (int i = 0; i < NTERM; ++i) {
      o0 += acc[i][0][r] * dinv_s[i][m];
      o1 += acc[i][1][r] * dinv_s[i][m];
    }
    op[(size_t)m * CDIM + nl]      = (__bf16)o0;
    op[(size_t)m * CDIM + nl + 16] = (__bf16)o1;
  }
}

// ---------------------------------------------------------------------------
extern "C" void kernel_launch(void* const* d_in, const int* in_sizes, int n_in,
                              void* d_out, int out_size, void* d_ws, size_t ws_size,
                              hipStream_t stream) {
  (void)in_sizes; (void)n_in; (void)out_size; (void)ws_size;
  const float* x  = (const float*)d_in[0];
  const float* y  = (const float*)d_in[1];
  const float* Wq = (const float*)d_in[2];
  const float* bq = (const float*)d_in[3];
  const float* Wk = (const float*)d_in[4];
  const float* bk = (const float*)d_in[5];
  const float* Wv = (const float*)d_in[6];
  const float* bv = (const float*)d_in[7];
  const float* Wp = (const float*)d_in[8];
  const float* bp = (const float*)d_in[9];
  float* out = (float*)d_out;

  char* ws = (char*)d_ws;
  size_t off = 0;
  const size_t MQ = (size_t)BDIM * NDIM * CDIM;  // == B*T*C
  __bf16* qbuf = (__bf16*)(ws + off); off += MQ * 2;
  __bf16* kbuf = (__bf16*)(ws + off); off += (size_t)NTERM * MQ * 2;
  __bf16* vbuf = (__bf16*)(ws + off); off += (size_t)NTERM * MQ * 2;
  __bf16* abuf = (__bf16*)(ws + off); off += MQ * 2;
  float* ksumb = (float*)(ws + off);  off += (size_t)NTERM * BDIM * CDIM * 4;
  float* ctxb  = (float*)(ws + off);  off += (size_t)NTERM * BDIM * HDIM * DHEAD * DHEAD * 4;

  dim3 blk(256);
  dim3 gproj(BDIM * NDIM / 128, CDIM / 64);

  // q = softmax_d(x @ Wq^T + bq), bf16
  proj_kernel<1, false><<<gproj, blk, 0, stream>>>(x, Wq, bq, qbuf, BDIM * NDIM, CDIM, CDIM);
  for (int i = 0; i < NTERM; ++i) {
    const float* yi = y + (size_t)i * MQ;
    proj_kernel<1, false><<<gproj, blk, 0, stream>>>(
        yi, Wk + (size_t)i * CDIM * CDIM, bk + i * CDIM,
        kbuf + (size_t)i * MQ, BDIM * TDIM, CDIM, CDIM);
    proj_kernel<0, false><<<gproj, blk, 0, stream>>>(
        yi, Wv + (size_t)i * CDIM * CDIM, bv + i * CDIM,
        vbuf + (size_t)i * MQ, BDIM * TDIM, CDIM, CDIM);
  }
  ksum_kernel<<<NTERM * BDIM * CDIM / 256, blk, 0, stream>>>(kbuf, ksumb);
  ctx_kernel<<<NTERM * BDIM * HDIM, blk, 0, stream>>>(kbuf, vbuf, ctxb);
  attn_kernel<<<dim3(NDIM / 64, BDIM * HDIM), blk, 0, stream>>>(qbuf, ctxb, ksumb, abuf);
  // final projection, fp32 out
  proj_kernel<2, true><<<gproj, blk, 0, stream>>>(abuf, Wp, bp, out, BDIM * NDIM, CDIM, CDIM);
}
